// TransformerBlock_8340826489276
// MI455X (gfx1250) — matrix operand order repaired and checked by reading, not compiled, not run on hardware
//
#include <hip/hip_runtime.h>
#include <hip/hip_bf16.h>
#include <math.h>

typedef __attribute__((ext_vector_type(16))) _Float16 v16h;
typedef __attribute__((ext_vector_type(8)))  float    v8f;

#define T_SEQ 2048
#define D_MODEL 1024
#define N_HEADS 16
#define HEAD_DIM 64
#define D_FF 4096
#define BATCH 2
#define M_ROWS (BATCH * T_SEQ)   // 4096

// ---- TDM availability probe (this toolchain: 6-arg builtin) ----------------
#if defined(__has_builtin)
#  if __has_builtin(__builtin_amdgcn_tensor_load_to_lds) && \
      __has_builtin(__builtin_amdgcn_s_wait_tensorcnt)
#    define USE_TDM 1
#  endif
#endif

__device__ __forceinline__ v8f wmma_f16(v16h a, v16h b, v8f c) {
    return __builtin_amdgcn_wmma_f32_16x16x32_f16(
        false, a, false, b, (short)0, c, false, false);
}

// Escape an LDS pointer into asm so the compiler must assume the buffer is
// written externally (TDM DMA writes LDS without any visible stores).
__device__ __forceinline__ void lds_escape(const void* p) {
    asm volatile("" : : "v"(p) : "memory");
}

// A-fragment (16x32 f16): lane l holds row M=l%16; element i holds
// K = (i<8 ? i : i+8) + 8*(l/16).  Two contiguous 16B runs -> 2x ds_load_b128.
__device__ __forceinline__ v16h load_a_frag(const _Float16* Asrow, int half) {
    v16h a;
#pragma unroll
    for (int i = 0; i < 16; ++i) {
        int k = ((i < 8) ? i : (i + 8)) + 8 * half;
        a[i] = Asrow[k];
    }
    return a;
}

#ifdef USE_TDM
typedef unsigned int u32x4 __attribute__((ext_vector_type(4)));
typedef int          i32x4 __attribute__((ext_vector_type(4)));
typedef int          i32x8 __attribute__((ext_vector_type(8)));

// Issue one 2D TDM tile load (f16 elements): tile_h rows x tile_w elems,
// row stride = stride_elems, into LDS byte offset lds_off.
__device__ __forceinline__ void tdm_load_2d(const _Float16* gptr,
                                            unsigned lds_off,
                                            unsigned tile_w, unsigned tile_h,
                                            unsigned stride_elems) {
    unsigned long long ga = (unsigned long long)(uintptr_t)gptr;
    u32x4 g0;
    g0[0] = 1u;                                        // count=1, user desc
    g0[1] = lds_off;                                   // lds_addr
    g0[2] = (unsigned)(ga & 0xFFFFFFFFu);              // global_addr lo
    g0[3] = (unsigned)((ga >> 32) & 0x01FFFFFFu) | (2u << 30);  // hi | type=2
    i32x8 g1;
    g1[0] = (int)(1u << 16);                           // data_size=1 -> 2B
    g1[1] = (int)((tile_w & 0xFFFFu) << 16);           // tensor_dim0[15:0]
    g1[2] = (int)((tile_w >> 16) | ((tile_h & 0xFFFFu) << 16));  // td0 hi|td1 lo
    g1[3] = (int)((tile_h >> 16) | (tile_w << 16));    // td1 hi | tile_dim0
    g1[4] = (int)tile_h;                               // tile_dim1 (tile_dim2=0)
    g1[5] = (int)stride_elems;                         // tensor_dim0_stride lo32
    g1[6] = 0;
    g1[7] = 0;
    i32x4 z4 = {};
    i32x8 z8 = {};
    __builtin_amdgcn_tensor_load_to_lds(g0, g1, z4, z4, z8, 0);
}
#endif

// ---------------------------------------------------------------------------
// fp32 -> f16 convert
// ---------------------------------------------------------------------------
__global__ __launch_bounds__(256) void cvt_f32_f16(const float* __restrict__ in,
                                                   _Float16* __restrict__ out, int n) {
    int i = blockIdx.x * 256 + threadIdx.x;
    if (i < n) out[i] = (_Float16)in[i];
}

// ---------------------------------------------------------------------------
// LayerNorm over C=1024, one block per row, f16 output
// ---------------------------------------------------------------------------
__global__ __launch_bounds__(256) void layernorm_f16(const float* __restrict__ x,
                                                     const float* __restrict__ g,
                                                     const float* __restrict__ b,
                                                     _Float16* __restrict__ out) {
    __shared__ float sh[256], sh2[256];
    int row = blockIdx.x;
    const float* xr = x + (size_t)row * D_MODEL;
    float s = 0.f, s2 = 0.f;
#pragma unroll
    for (int i = 0; i < 4; ++i) {
        float v = xr[threadIdx.x + i * 256];
        s += v; s2 += v * v;
    }
    sh[threadIdx.x] = s; sh2[threadIdx.x] = s2;
    __syncthreads();
    for (int off = 128; off > 0; off >>= 1) {
        if (threadIdx.x < off) {
            sh[threadIdx.x]  += sh[threadIdx.x + off];
            sh2[threadIdx.x] += sh2[threadIdx.x + off];
        }
        __syncthreads();
    }
    float mu  = sh[0] * (1.0f / D_MODEL);
    float var = sh2[0] * (1.0f / D_MODEL) - mu * mu;
    float rs  = rsqrtf(var + 1e-5f);
#pragma unroll
    for (int i = 0; i < 4; ++i) {
        int c = threadIdx.x + i * 256;
        out[(size_t)row * D_MODEL + c] = (_Float16)((xr[c] - mu) * rs * g[c] + b[c]);
    }
}

// ---------------------------------------------------------------------------
// WMMA GEMM: out[M,N] = A[M,K](f16) B[K,N](f16) + bias (+resid) (gelu?)
// block 256 thr = 8 waves (4M x 2N); block tile 128x128; wave tile 32x64;
// K-step 32.  TDM double-buffered LDS when available, else reg-pipelined.
// ---------------------------------------------------------------------------
#define BM 128
#define BN 128
#define BK 32
#define A_ELEMS (BM * BK)              // 4096 halves = 8KB
#define B_ELEMS (BK * BN)              // 4096 halves = 8KB
#define BUF_ELEMS (A_ELEMS + B_ELEMS)  // 16KB
#ifdef USE_TDM
#define NBUF 2
#else
#define NBUF 1
#endif

__device__ __forceinline__ void epilogue_frag(v8f c, int row0, int col0, int half, int mrow,
                                              const float* __restrict__ bias,
                                              const float* __restrict__ resid,
                                              float* __restrict__ outF,
                                              _Float16* __restrict__ outH,
                                              int N, int act) {
#pragma unroll
    for (int v = 0; v < 8; ++v) {
        int row = row0 + v + 8 * half;
        int col = col0 + mrow;
        float val = c[v] + bias[col];
        if (resid) val += resid[(size_t)row * N + col];
        if (act) val = 0.5f * val * (1.0f + erff(val * 0.70710678118f));
        if (outF) outF[(size_t)row * N + col] = val;
        if (outH) outH[(size_t)row * N + col] = (_Float16)val;
    }
}

__global__ __launch_bounds__(256) void gemm_wmma(const _Float16* __restrict__ A,
                                                 const _Float16* __restrict__ B,
                                                 const float* __restrict__ bias,
                                                 const float* __restrict__ resid,
                                                 float* __restrict__ outF,
                                                 _Float16* __restrict__ outH,
                                                 int M, int N, int K, int act) {
    __shared__ alignas(128) _Float16 smem[NBUF][BUF_ELEMS];

    const int tid  = threadIdx.x;
    const int w    = tid >> 5;
    const int lane = tid & 31;
    const int half = lane >> 4;
    const int mrow = lane & 15;
    const int wm   = w & 3;    // M sub-tile of 32
    const int wn   = w >> 2;   // N sub-tile of 64
    const int bm0  = blockIdx.y * BM;
    const int bn0  = blockIdx.x * BN;
    const int nk   = K / BK;

    v8f zero = {};
    v8f c00 = zero, c01 = zero, c02 = zero, c03 = zero;
    v8f c10 = zero, c11 = zero, c12 = zero, c13 = zero;

    lds_escape(&smem[0][0]);  // address escapes: TDM writes are now visible

#ifdef USE_TDM
    // prologue: DMA tile 0 into buffer 0 (wave 0 issues; EXEC-independent DMA)
    if (w == 0) {
        tdm_load_2d(A + (size_t)bm0 * K, 0, BK, BM, K);
        tdm_load_2d(B + bn0, A_ELEMS * 2, BN, BK, N);
    }
    for (int t = 0; t < nk; ++t) {
        const int buf = t & 1;
        if (w == 0) {
            if (t + 1 < nk) {
                const int k0n = (t + 1) * BK;
                const unsigned lb = (unsigned)((buf ^ 1) * BUF_ELEMS * 2);
                tdm_load_2d(A + (size_t)bm0 * K + k0n, lb, BK, BM, K);
                tdm_load_2d(B + (size_t)k0n * N + bn0, lb + A_ELEMS * 2, BN, BK, N);
                __builtin_amdgcn_s_wait_tensorcnt(2);  // tile t landed
            } else {
                __builtin_amdgcn_s_wait_tensorcnt(0);
            }
        }
        __syncthreads();
        asm volatile("" ::: "memory");  // order LDS reads after the DMA wait
        const _Float16* Abuf = &smem[buf][0];
        const _Float16* Bbuf = &smem[buf][A_ELEMS];
#else
    uint4 ra[2], rb[2];
    auto load_regs = [&](int k0) {
#pragma unroll
        for (int r = 0; r < 2; ++r) {
            int seg = tid + r * 256;  // 0..511
            ra[r] = *(const uint4*)&A[(size_t)(bm0 + (seg >> 2)) * K + k0 + (seg & 3) * 8];
            rb[r] = *(const uint4*)&B[(size_t)(k0 + (seg >> 4)) * N + bn0 + (seg & 15) * 8];
        }
    };
    load_regs(0);
    for (int t = 0; t < nk; ++t) {
#pragma unroll
        for (int r = 0; r < 2; ++r) {
            int seg = tid + r * 256;
            *(uint4*)&smem[0][(seg >> 2) * BK + (seg & 3) * 8]           = ra[r];
            *(uint4*)&smem[0][A_ELEMS + (seg >> 4) * BN + (seg & 15) * 8] = rb[r];
        }
        __syncthreads();
        if (t + 1 < nk) load_regs((t + 1) * BK);  // overlap with compute
        const _Float16* Abuf = &smem[0][0];
        const _Float16* Bbuf = &smem[0][A_ELEMS];
#endif
        v16h a0 = load_a_frag(Abuf + (wm * 32 + mrow) * BK, half);
        v16h a1 = load_a_frag(Abuf + (wm * 32 + 16 + mrow) * BK, half);
        v16h b0, b1, b2, b3;
#pragma unroll
        for (int i = 0; i < 16; ++i) {
            int k = ((i < 8) ? i : (i + 8)) + 8 * half;
            const _Float16* br = Bbuf + k * BN + wn * 64 + mrow;
            b0[i] = br[0]; b1[i] = br[16]; b2[i] = br[32]; b3[i] = br[48];
        }
        c00 = wmma_f16(a0, b0, c00);
        c01 = wmma_f16(a0, b1, c01);
        c02 = wmma_f16(a0, b2, c02);
        c03 = wmma_f16(a0, b3, c03);
        c10 = wmma_f16(a1, b0, c10);
        c11 = wmma_f16(a1, b1, c11);
        c12 = wmma_f16(a1, b2, c12);
        c13 = wmma_f16(a1, b3, c13);
        __syncthreads();
    }

    const int rb0 = bm0 + wm * 32;
    const int cb0 = bn0 + wn * 64;
    epilogue_frag(c00, rb0,      cb0,      half, mrow, bias, resid, outF, outH, N, act);
    epilogue_frag(c01, rb0,      cb0 + 16, half, mrow, bias, resid, outF, outH, N, act);
    epilogue_frag(c02, rb0,      cb0 + 32, half, mrow, bias, resid, outF, outH, N, act);
    epilogue_frag(c03, rb0,      cb0 + 48, half, mrow, bias, resid, outF, outH, N, act);
    epilogue_frag(c10, rb0 + 16, cb0,      half, mrow, bias, resid, outF, outH, N, act);
    epilogue_frag(c11, rb0 + 16, cb0 + 16, half, mrow, bias, resid, outF, outH, N, act);
    epilogue_frag(c12, rb0 + 16, cb0 + 32, half, mrow, bias, resid, outF, outH, N, act);
    epilogue_frag(c13, rb0 + 16, cb0 + 48, half, mrow, bias, resid, outF, outH, N, act);
}

// ---------------------------------------------------------------------------
// Causal flash attention. qkv is f16 (M_ROWS x 3072), cols [q|k|v] each 1024.
// grid (B*H, T/128); block 256 = 8 waves; wave owns 16 query rows.
// ---------------------------------------------------------------------------
__global__ __launch_bounds__(256, 2) void attn_flash(const _Float16* __restrict__ qkv,
                                                     _Float16* __restrict__ attn_out) {
    __shared__ alignas(32) _Float16 Kt[64][32];      // [d][key] (transposed)
    __shared__ alignas(32) _Float16 Vs[32][64];      // [key][d]
    __shared__ alignas(32) _Float16 Ps[8][16][32];   // per-wave P tile

    const int tid  = threadIdx.x;
    const int w    = tid >> 5;
    const int lane = tid & 31;
    const int half = lane >> 4;
    const int mrow = lane & 15;
    const int bh   = blockIdx.x;
    const int b    = bh >> 4;
    const int h    = bh & 15;
    const int band = blockIdx.y * 128;
    const int t0   = band + w * 16;
    const int gr   = b * T_SEQ;

    v16h qf[2];
#pragma unroll
    for (int c = 0; c < 2; ++c)
#pragma unroll
        for (int i = 0; i < 16; ++i) {
            int d = ((i < 8) ? i : (i + 8)) + 8 * half + c * 32;
            float qv = (float)qkv[(size_t)(gr + t0 + mrow) * 3072 + h * 64 + d];
            qf[c][i] = (_Float16)(qv * 0.125f);  // 1/sqrt(64)
        }

    float mr[8], lr[8];
#pragma unroll
    for (int v = 0; v < 8; ++v) { mr[v] = -1e30f; lr[v] = 0.f; }
    v8f o0 = {}, o1 = {}, o2 = {}, o3 = {};

    const int nEnd = band + 128;
    for (int n0 = 0; n0 < nEnd; n0 += 32) {
        __syncthreads();
        {
            int key   = tid >> 3;
            int dbase = (tid & 7) * 8;
            const _Float16* ksrc =
                &qkv[(size_t)(gr + n0 + key) * 3072 + 1024 + h * 64 + dbase];
#pragma unroll
            for (int j = 0; j < 8; ++j) Kt[dbase + j][key] = ksrc[j];
            *(uint4*)&Vs[key][dbase] =
                *(const uint4*)&qkv[(size_t)(gr + n0 + key) * 3072 + 2048 + h * 64 + dbase];
        }
        __syncthreads();

        if (n0 <= t0 + 15) {
            v8f s0 = {}, s1 = {};
#pragma unroll
            for (int c = 0; c < 2; ++c) {
                v16h kb0, kb1;
#pragma unroll
                for (int i = 0; i < 16; ++i) {
                    int k = ((i < 8) ? i : (i + 8)) + 8 * half;
                    kb0[i] = Kt[c * 32 + k][mrow];
                    kb1[i] = Kt[c * 32 + k][16 + mrow];
                }
                s0 = wmma_f16(qf[c], kb0, s0);
                s1 = wmma_f16(qf[c], kb1, s1);
            }
#pragma unroll
            for (int v = 0; v < 8; ++v) {
                int grow = t0 + v + 8 * half;
                float a0 = ((n0 + mrow)      <= grow) ? s0[v] : -1e30f;
                float a1 = ((n0 + 16 + mrow) <= grow) ? s1[v] : -1e30f;
                float mx = fmaxf(a0, a1);
                mx = fmaxf(mx, __shfl_xor(mx, 1));
                mx = fmaxf(mx, __shfl_xor(mx, 2));
                mx = fmaxf(mx, __shfl_xor(mx, 4));
                mx = fmaxf(mx, __shfl_xor(mx, 8));
                float newm = fmaxf(mr[v], mx);
                float p0 = expf(a0 - newm);
                float p1 = expf(a1 - newm);
                float corr = expf(mr[v] - newm);
                float rs = p0 + p1;
                rs += __shfl_xor(rs, 1);
                rs += __shfl_xor(rs, 2);
                rs += __shfl_xor(rs, 4);
                rs += __shfl_xor(rs, 8);
                lr[v] = lr[v] * corr + rs;
                mr[v] = newm;
                o0[v] *= corr; o1[v] *= corr; o2[v] *= corr; o3[v] *= corr;
                int Mr = v + 8 * half;
                Ps[w][Mr][mrow]      = (_Float16)p0;
                Ps[w][Mr][16 + mrow] = (_Float16)p1;
            }
            asm volatile("s_wait_dscnt 0" ::: "memory");
            v16h pa;
#pragma unroll
            for (int i = 0; i < 16; ++i) {
                int k = ((i < 8) ? i : (i + 8)) + 8 * half;
                pa[i] = Ps[w][mrow][k];
            }
            v16h vb0, vb1, vb2, vb3;
#pragma unroll
            for (int i = 0; i < 16; ++i) {
                int k = ((i < 8) ? i : (i + 8)) + 8 * half;
                vb0[i] = Vs[k][mrow]; vb1[i] = Vs[k][16 + mrow];
                vb2[i] = Vs[k][32 + mrow]; vb3[i] = Vs[k][48 + mrow];
            }
            o0 = wmma_f16(pa, vb0, o0);
            o1 = wmma_f16(pa, vb1, o1);
            o2 = wmma_f16(pa, vb2, o2);
            o3 = wmma_f16(pa, vb3, o3);
        }
    }

#pragma unroll
    for (int v = 0; v < 8; ++v) {
        float inv = 1.0f / lr[v];
        int t = t0 + v + 8 * half;
        size_t base = (size_t)(gr + t) * D_MODEL + h * 64;
        attn_out[base + 0 * 16 + mrow] = (_Float16)(o0[v] * inv);
        attn_out[base + 1 * 16 + mrow] = (_Float16)(o1[v] * inv);
        attn_out[base + 2 * 16 + mrow] = (_Float16)(o2[v] * inv);
        attn_out[base + 3 * 16 + mrow] = (_Float16)(o3[v] * inv);
    }
}

// ---------------------------------------------------------------------------
// Host launcher
// ---------------------------------------------------------------------------
extern "C" void kernel_launch(void* const* d_in, const int* in_sizes, int n_in,
                              void* d_out, int out_size, void* d_ws, size_t ws_size,
                              hipStream_t stream) {
    (void)in_sizes; (void)n_in; (void)out_size; (void)ws_size;

    const float* x     = (const float*)d_in[0];
    const float* ln1_g = (const float*)d_in[1];
    const float* ln1_b = (const float*)d_in[2];
    const float* w_qkv = (const float*)d_in[3];
    const float* b_qkv = (const float*)d_in[4];
    const float* w_out = (const float*)d_in[5];
    const float* b_out = (const float*)d_in[6];
    const float* ln2_g = (const float*)d_in[7];
    const float* ln2_b = (const float*)d_in[8];
    const float* w_ff1 = (const float*)d_in[9];
    const float* b_ff1 = (const float*)d_in[10];
    const float* w_ff2 = (const float*)d_in[11];
    const float* b_ff2 = (const float*)d_in[12];

    char* p = (char*)d_ws;
    _Float16* wq_h   = (_Float16*)p; p += (size_t)D_MODEL * 3 * D_MODEL * 2;
    _Float16* wo_h   = (_Float16*)p; p += (size_t)D_MODEL * D_MODEL * 2;
    _Float16* wf1_h  = (_Float16*)p; p += (size_t)D_MODEL * D_FF * 2;
    _Float16* wf2_h  = (_Float16*)p; p += (size_t)D_FF * D_MODEL * 2;
    _Float16* h1_h   = (_Float16*)p; p += (size_t)M_ROWS * D_MODEL * 2;
    _Float16* qkv_h  = (_Float16*)p; p += (size_t)M_ROWS * 3 * D_MODEL * 2;
    _Float16* attn_h = (_Float16*)p; p += (size_t)M_ROWS * D_MODEL * 2;
    float*    x1     = (float*)p;    p += (size_t)M_ROWS * D_MODEL * 4;
    _Float16* h2_h   = (_Float16*)p; p += (size_t)M_ROWS * D_MODEL * 2;
    _Float16* ff_h   = (_Float16*)p; p += (size_t)M_ROWS * D_FF * 2;

    auto cvt = [&](const float* src, _Float16* dst, int n) {
        cvt_f32_f16<<<(n + 255) / 256, 256, 0, stream>>>(src, dst, n);
    };
    cvt(w_qkv, wq_h,  D_MODEL * 3 * D_MODEL);
    cvt(w_out, wo_h,  D_MODEL * D_MODEL);
    cvt(w_ff1, wf1_h, D_MODEL * D_FF);
    cvt(w_ff2, wf2_h, D_FF * D_MODEL);

    layernorm_f16<<<M_ROWS, 256, 0, stream>>>(x, ln1_g, ln1_b, h1_h);

    gemm_wmma<<<dim3(3 * D_MODEL / BN, M_ROWS / BM), 256, 0, stream>>>(
        h1_h, wq_h, b_qkv, nullptr, nullptr, qkv_h,
        M_ROWS, 3 * D_MODEL, D_MODEL, 0);

    attn_flash<<<dim3(BATCH * N_HEADS, T_SEQ / 128), 256, 0, stream>>>(qkv_h, attn_h);

    gemm_wmma<<<dim3(D_MODEL / BN, M_ROWS / BM), 256, 0, stream>>>(
        attn_h, wo_h, b_out, x, x1, nullptr,
        M_ROWS, D_MODEL, D_MODEL, 0);

    layernorm_f16<<<M_ROWS, 256, 0, stream>>>(x1, ln2_g, ln2_b, h2_h);

    gemm_wmma<<<dim3(D_FF / BN, M_ROWS / BM), 256, 0, stream>>>(
        h2_h, wf1_h, b_ff1, nullptr, nullptr, ff_h,
        M_ROWS, D_FF, D_MODEL, 1);

    gemm_wmma<<<dim3(D_MODEL / BN, M_ROWS / BM), 256, 0, stream>>>(
        ff_h, wf2_h, b_ff2, x1, (float*)d_out, nullptr,
        M_ROWS, D_MODEL, D_FF, 0);
}
